// Binary_CNN2_58892591563202
// MI455X (gfx1250) — 8-bit matrix operand order repaired and checked by reading, compiled, not run on hardware
//
#include <hip/hip_runtime.h>
#include <hip/hip_bf16.h>

typedef int   v16i __attribute__((ext_vector_type(16)));
typedef float v8f  __attribute__((ext_vector_type(8)));

#define BB 4096
#define F1 6272
#define HH 2048
#define NC 10
#define EPS 1e-5f

#define FP8_P1 ((unsigned char)0x38)
#define FP8_M1 ((unsigned char)0xB8)

__global__ void __launch_bounds__(256)
binarize_w2_kernel(const float* __restrict__ W2, unsigned char* __restrict__ Wb2, int n4) {
    int i = blockIdx.x * blockDim.x + threadIdx.x;
    if (i >= n4) return;
    float4 w = *(const float4*)(W2 + (size_t)i * 4);
    union { unsigned char c[4]; int u; } p;
    p.c[0] = (w.x >= 0.f) ? FP8_P1 : FP8_M1;
    p.c[1] = (w.y >= 0.f) ? FP8_P1 : FP8_M1;
    p.c[2] = (w.z >= 0.f) ? FP8_P1 : FP8_M1;
    p.c[3] = (w.w >= 0.f) ? FP8_P1 : FP8_M1;
    ((int*)Wb2)[i] = p.u;
}

__global__ void __launch_bounds__(256)
conv_pool_bn_sign_kernel(const float* __restrict__ x,
                         const float* __restrict__ W1,
                         const float* __restrict__ b1,
                         const float* __restrict__ g1,
                         const float* __restrict__ be1,
                         const float* __restrict__ m1,
                         const float* __restrict__ v1,
                         unsigned char* __restrict__ A) {
    int idx = blockIdx.x * blockDim.x + threadIdx.x;
    int pw = idx % 14;  int t = idx / 14;
    int ph = t % 14;    t /= 14;
    int c  = t % 32;    int n = t / 32;
    if (n >= BB) return;

    float ws[9];
#pragma unroll
    for (int k = 0; k < 9; ++k)
        ws[k] = (W1[c * 9 + k] >= 0.f) ? 1.f : -1.f;

    const float* xn = x + (size_t)n * 784;
    float xs[4][4];
#pragma unroll
    for (int i = 0; i < 4; ++i) {
        int h = 2 * ph - 1 + i;
#pragma unroll
        for (int j = 0; j < 4; ++j) {
            int w = 2 * pw - 1 + j;
            float v = 0.f;
            if (h >= 0 && h < 28 && w >= 0 && w < 28)
                v = (xn[h * 28 + w] >= 0.f) ? 1.f : -1.f;
            xs[i][j] = v;
        }
    }

    float bias = b1[c];
    float mx = -INFINITY;
#pragma unroll
    for (int dy = 0; dy < 2; ++dy)
#pragma unroll
        for (int dx = 0; dx < 2; ++dx) {
            float s = bias;
#pragma unroll
            for (int i = 0; i < 3; ++i)
#pragma unroll
                for (int j = 0; j < 3; ++j)
                    s += xs[dy + i][dx + j] * ws[i * 3 + j];
            mx = fmaxf(mx, s);
        }

    float z = g1[c] * (mx - m1[c]) * rsqrtf(v1[c] + EPS) + be1[c];
    A[(size_t)n * F1 + c * 196 + ph * 14 + pw] = (z >= 0.f) ? FP8_P1 : FP8_M1;
}

__global__ void __launch_bounds__(128)
bin_gemm_wmma_kernel(const unsigned char* __restrict__ A,
                     const unsigned char* __restrict__ Bw,
                     const float* __restrict__ b2,
                     const float* __restrict__ g2,
                     const float* __restrict__ be2,
                     const float* __restrict__ m2,
                     const float* __restrict__ v2,
                     float* __restrict__ Z) {
    const int lane = threadIdx.x & 31;
    const int wave = threadIdx.x >> 5;
    const int wy = wave >> 1, wx = wave & 1;
    const int mBase = blockIdx.x * 128 + wy * 64;
    const int nBase = blockIdx.y * 128 + wx * 64;
    const int hlf = lane >> 4;
    const int l16 = lane & 15;

    const unsigned char* aRow[4];
    const unsigned char* bRow[4];
#pragma unroll
    for (int i = 0; i < 4; ++i) {
        aRow[i] = A  + (size_t)(mBase + i * 16 + l16) * F1 + hlf * 16;
        bRow[i] = Bw + (size_t)(nBase + i * 16 + l16) * F1 + hlf * 16;
    }

    v8f acc[4][4] = {};

#pragma unroll 1
    for (int kb = 0; kb < F1; kb += 128) {
        v16i a[4], b[4];
#pragma unroll
        for (int i = 0; i < 4; ++i) {
#pragma unroll
            for (int blk = 0; blk < 2; ++blk)
#pragma unroll
                for (int ci = 0; ci < 4; ++ci) {
                    int2 q = *(const int2*)(aRow[i] + kb + blk * 64 + (ci >> 1) * 32 + (ci & 1) * 8);
                    a[i][blk * 8 + 2 * ci]     = q.x;
                    a[i][blk * 8 + 2 * ci + 1] = q.y;
                }
        }
#pragma unroll
        for (int j = 0; j < 4; ++j) {
#pragma unroll
            for (int ch = 0; ch < 4; ++ch) {
                int4 q = *(const int4*)(bRow[j] + kb + ch * 32);
                b[j][4 * ch]     = q.x; b[j][4 * ch + 1] = q.y;
                b[j][4 * ch + 2] = q.z; b[j][4 * ch + 3] = q.w;
            }
        }
#pragma unroll
        for (int i = 0; i < 4; ++i)
#pragma unroll
            for (int j = 0; j < 4; ++j)
                acc[i][j] = __builtin_amdgcn_wmma_f32_16x16x128_fp8_fp8(
                    a[i], b[j], (short)0, acc[i][j], false, false);
    }

#pragma unroll
    for (int j = 0; j < 4; ++j) {
        const int n = nBase + j * 16 + l16;
        const float bb = b2[n], gg = g2[n], mm = m2[n], bbe = be2[n];
        const float rs = rsqrtf(v2[n] + EPS);
#pragma unroll
        for (int i = 0; i < 4; ++i) {
            const int mrow = mBase + i * 16 + hlf * 8;
#pragma unroll
            for (int r = 0; r < 8; ++r) {
                float z = acc[i][j][r] + bb;
                z = gg * (z - mm) * rs + bbe;
                z = fminf(1.f, fmaxf(-1.f, z));
                Z[(size_t)(mrow + r) * HH + n] = z;
            }
        }
    }
}

__global__ void __launch_bounds__(256)
head_logsoftmax_kernel(const float* __restrict__ Z,
                       const float* __restrict__ W3,
                       const float* __restrict__ b3,
                       float* __restrict__ out) {
    const int row  = blockIdx.x * 8 + (threadIdx.x >> 5);
    const int lane = threadIdx.x & 31;
    const float* zr = Z + (size_t)row * HH;

    float acc[NC];
#pragma unroll
    for (int c = 0; c < NC; ++c) acc[c] = 0.f;

    for (int k = lane * 4; k < HH; k += 32 * 4) {
        float4 zv = *(const float4*)(zr + k);
#pragma unroll
        for (int c = 0; c < NC; ++c) {
            float4 wv = *(const float4*)(W3 + (size_t)c * HH + k);
            acc[c] += zv.x * wv.x + zv.y * wv.y + zv.z * wv.z + zv.w * wv.w;
        }
    }

#pragma unroll
    for (int c = 0; c < NC; ++c)
#pragma unroll
        for (int off = 16; off > 0; off >>= 1)
            acc[c] += __shfl_xor(acc[c], off, 32);

    float logit[NC], mx = -INFINITY;
#pragma unroll
    for (int c = 0; c < NC; ++c) {
        logit[c] = acc[c] + b3[c];
        mx = fmaxf(mx, logit[c]);
    }
    float s = 0.f;
#pragma unroll
    for (int c = 0; c < NC; ++c) s += __expf(logit[c] - mx);
    const float lse = mx + __logf(s);

    if (lane < NC) out[(size_t)row * NC + lane] = logit[lane] - lse;
}

extern "C" void kernel_launch(void* const* d_in, const int* in_sizes, int n_in,
                              void* d_out, int out_size, void* d_ws, size_t ws_size,
                              hipStream_t stream) {
    const float* x   = (const float*)d_in[0];
    const float* W1  = (const float*)d_in[1];
    const float* b1  = (const float*)d_in[2];
    const float* g1  = (const float*)d_in[3];
    const float* be1 = (const float*)d_in[4];
    const float* m1  = (const float*)d_in[5];
    const float* v1  = (const float*)d_in[6];
    const float* W2  = (const float*)d_in[7];
    const float* b2  = (const float*)d_in[8];
    const float* g2  = (const float*)d_in[9];
    const float* be2 = (const float*)d_in[10];
    const float* m2  = (const float*)d_in[11];
    const float* v2  = (const float*)d_in[12];
    const float* W3  = (const float*)d_in[13];
    const float* b3  = (const float*)d_in[14];
    float* out = (float*)d_out;

    char* ws = (char*)d_ws;
    const size_t szWb2 = (size_t)HH * F1;
    const size_t szA   = (size_t)BB * F1;
    unsigned char* Wb2  = (unsigned char*)ws;
    unsigned char* Amat = (unsigned char*)(ws + szWb2);
    float*         Zbuf = (float*)(ws + szWb2 + szA);

    {
        int n4 = (HH * F1) / 4;
        binarize_w2_kernel<<<(n4 + 255) / 256, 256, 0, stream>>>(W2, Wb2, n4);
    }
    {
        int total = BB * 32 * 14 * 14;
        conv_pool_bn_sign_kernel<<<total / 256, 256, 0, stream>>>(
            x, W1, b1, g1, be1, m1, v1, Amat);
    }
    {
        dim3 grid(BB / 128, HH / 128);
        bin_gemm_wmma_kernel<<<grid, 128, 0, stream>>>(
            Amat, Wb2, b2, g2, be2, m2, v2, Zbuf);
    }
    {
        head_logsoftmax_kernel<<<BB / 8, 256, 0, stream>>>(Zbuf, W3, b3, out);
    }
}
